// SpectralAttention_41326175322587
// MI455X (gfx1250) — matrix operand order repaired and checked by reading, not compiled, not run on hardware
//
#include <hip/hip_runtime.h>

// ---------------------------------------------------------------------------
// SpectralAttention forward for MI455X (gfx1250, wave32, WMMA + async-LDS).
//
//   q,k,v  = x @ W^T + b          (WMMA bf16 GEMMs, async-LDS double buffered)
//   f      = sb @ td^T + b        (WMMA bf16 GEMM)
//   kn,vn  = l2norm rows (per head)
//   k_f,v_f= causal conv per (h,d) channel   (tiled LDS VALU FMA)
//   g_t    = relu(v_f^T (wgz .* kv) k_f + b)^2 + eps
//   sp[l]  = l2norm( sum_{t<=l} g_t (q[l].v_f[t]) k_f[t,:] )   <- causal attn
//            (g_scan denominator cancels inside l2norm)
//   out    = sp @ wo^T + b        (WMMA bf16 GEMM)
// ---------------------------------------------------------------------------

#define L_SEQ 2048
#define D_MOD 512
#define NHEAD 8
#define HDIM  64
#define EPSF  1e-5f

#ifndef USE_ASYNC_COPY
#define USE_ASYNC_COPY 1
#endif

typedef __attribute__((ext_vector_type(16))) __bf16 v16bf;
typedef __attribute__((ext_vector_type(8)))  float  v8f;

// 16-byte global -> LDS copy. Async path uses the gfx1250 ASYNCcnt-tracked
// direct-to-LDS instruction (no VGPR roundtrip); fallback is a sync copy.
static __device__ inline void cp16B(const __bf16* g, __bf16* l) {
#if USE_ASYNC_COPY
  unsigned int loff = (unsigned int)(unsigned long long)l;  // LDS byte offset
  unsigned long long ga = (unsigned long long)g;
  asm volatile("global_load_async_to_lds_b128 %0, %1, off"
               :: "v"(loff), "v"(ga) : "memory");
#else
  *(uint4*)l = *(const uint4*)g;
#endif
}

#if USE_ASYNC_COPY
#define ASYNC_WAIT(n) asm volatile("s_wait_asynccnt %0" :: "i"(n) : "memory")
#else
#define ASYNC_WAIT(n) ((void)0)
#endif

// ----- fp32 -> bf16 convert -------------------------------------------------
__global__ __launch_bounds__(256)
void cvt_f32_bf16(const float* __restrict__ in, __bf16* __restrict__ out, int n) {
  int i = blockIdx.x * 256 + threadIdx.x;
  if (i < n) out[i] = (__bf16)in[i];
}

// ----- C[M,N] = A[M,K] @ W[N,K]^T + bias  (bf16 WMMA, f32 accum) ------------
// block = 256 threads = 8 waves; block tile 64(M) x 128(N); K-step 64;
// double-buffered async-LDS staging; wave tile 16x64.
__global__ __launch_bounds__(256)
void gemm_bf16_wmma(const __bf16* __restrict__ A, const __bf16* __restrict__ W,
                    const float* __restrict__ bias, float* __restrict__ C,
                    int M, int N, int K) {
  __shared__ __bf16 As[2][64][72];    // padded rows: conflict-free b128 reads
  __shared__ __bf16 Bs[2][128][72];
  int wave = threadIdx.x >> 5;
  int lane = threadIdx.x & 31;
  int r = lane & 15, half = lane >> 4;
  int m_blk = blockIdx.y * 64, n_blk = blockIdx.x * 128;
  int msub = (wave >> 1) * 16;
  int nsub = (wave & 1) * 64;

  auto issue_tile = [&](int stage, int k0) {
    // A tile: 64 rows x 64 cols = 512 x 16B chunks -> 2 async ops / thread
    for (int c = threadIdx.x; c < 512; c += 256) {
      int row = c >> 3, ch = (c & 7) * 8;
      cp16B(A + (m_blk + row) * K + k0 + ch, &As[stage][row][ch]);
    }
    // B tile: 128 rows x 64 cols = 1024 chunks -> 4 async ops / thread
    for (int c = threadIdx.x; c < 1024; c += 256) {
      int row = c >> 3, ch = (c & 7) * 8;
      cp16B(W + (n_blk + row) * K + k0 + ch, &Bs[stage][row][ch]);
    }
  };

  v8f acc[4] = {};
  issue_tile(0, 0);
  int nk = K >> 6;
  for (int it = 0; it < nk; ++it) {
    int stage = it & 1;
    if (it + 1 < nk) {
      issue_tile(stage ^ 1, (it + 1) << 6);
      ASYNC_WAIT(6);                 // 6 newer ops outstanding -> stage done
    } else {
      ASYNC_WAIT(0);
    }
    __syncthreads();
#pragma unroll
    for (int kc = 0; kc < 2; ++kc) {
      // A fragment (16x32): lane r<16 -> M=r, K {0..7,16..23}; r>=16 -> {8..15,24..31}
      v16bf a;
      {
        int cb = kc * 32 + half * 8;
#pragma unroll
        for (int i = 0; i < 8; ++i) a[i] = As[stage][msub + r][cb + i];
#pragma unroll
        for (int i = 0; i < 8; ++i) a[8 + i] = As[stage][msub + r][cb + 16 + i];
      }
#pragma unroll
      for (int n = 0; n < 4; ++n) {
        // B fragment: lane n<16 -> W row n, K {0..7,16..23}; lane>=16 -> {8..15,24..31}
        v16bf b;
        int brow = nsub + 16 * n + r;
        int bc = kc * 32 + half * 8;
#pragma unroll
        for (int i = 0; i < 16; ++i) b[i] = Bs[stage][brow][bc + (i < 8 ? i : i + 8)];
        acc[n] = __builtin_amdgcn_wmma_f32_16x16x32_bf16(
            false, a, false, b, (short)0, acc[n], false, false);
      }
    }
    __syncthreads();
  }
  // C f32 layout: VGPR j -> M = j + 8*(lane>=16), N = lane&15
#pragma unroll
  for (int n = 0; n < 4; ++n) {
    int col = n_blk + nsub + 16 * n + r;
    float bv = bias[col];
#pragma unroll
    for (int j = 0; j < 8; ++j) {
      int row = m_blk + msub + j + half * 8;
      C[row * N + col] = acc[n][j] + bv;
    }
  }
}

// ----- per-(l,h) L2 row norm of k and v (64-wide rows) ----------------------
__global__ __launch_bounds__(256)
void l2norm_rows(const float* __restrict__ k, const float* __restrict__ v,
                 float* __restrict__ kn, float* __restrict__ vn) {
  int wave = threadIdx.x >> 5, lane = threadIdx.x & 31;
  int row = blockIdx.x * 8 + wave;
  const float* pk = k + row * 64;
  const float* pv = v + row * 64;
  float k0 = pk[lane], k1 = pk[lane + 32];
  float v0 = pv[lane], v1 = pv[lane + 32];
  float sk = k0 * k0 + k1 * k1;
  float sv = v0 * v0 + v1 * v1;
#pragma unroll
  for (int m = 1; m <= 16; m <<= 1) {
    sk += __shfl_xor(sk, m, 32);
    sv += __shfl_xor(sv, m, 32);
  }
  float ck = 1.0f / fmaxf(sqrtf(sk), EPSF);
  float cv = 1.0f / fmaxf(sqrtf(sv), EPSF);
  kn[row * 64 + lane] = k0 * ck;  kn[row * 64 + lane + 32] = k1 * ck;
  vn[row * 64 + lane] = v0 * cv;  vn[row * 64 + lane + 32] = v1 * cv;
}

// ----- causal conv per (h,d):  out[l,d] = sum_{t<=l} in[t,d] * f[l-t,d] -----
// Writes fp32 (for gates) and bf16 (for attention tiles).
__global__ __launch_bounds__(256)
void conv_causal(const float* __restrict__ kn, const float* __restrict__ vn,
                 const float* __restrict__ flt,
                 float* __restrict__ kf, float* __restrict__ vf,
                 __bf16* __restrict__ kfb, __bf16* __restrict__ vfb) {
  __shared__ float ks[32][64];
  __shared__ float vs[32][64];
  __shared__ float fs[63][64];
  int h = blockIdx.y;
  int l0 = blockIdx.x * 32;
  int d = threadIdx.x & 63;
  int lq = threadIdx.x >> 6;
  float accK[8] = {}, accV[8] = {};
  for (int t0 = 0; t0 <= l0; t0 += 32) {
    for (int i = threadIdx.x; i < 32 * 64; i += 256) {
      int t = i >> 6, dd = i & 63;
      ks[t][dd] = kn[(t0 + t) * D_MOD + h * HDIM + dd];
      vs[t][dd] = vn[(t0 + t) * D_MOD + h * HDIM + dd];
    }
    for (int i = threadIdx.x; i < 63 * 64; i += 256) {
      int idx = i >> 6, dd = i & 63;
      int lag = l0 - t0 - 31 + idx;          // f[lag<0] == 0 -> causality
      fs[idx][dd] = (lag >= 0) ? flt[lag * D_MOD + h * HDIM + dd] : 0.0f;
    }
    if (t0 + 32 <= l0)
      __builtin_prefetch(kn + (t0 + 32) * D_MOD + h * HDIM, 0, 1);
    __syncthreads();
#pragma unroll
    for (int j = 0; j < 8; ++j) {
      int lrel = lq * 8 + j;
#pragma unroll
      for (int tt = 0; tt < 32; ++tt) {
        float f = fs[lrel + 31 - tt][d];
        accK[j] = fmaf(ks[tt][d], f, accK[j]);
        accV[j] = fmaf(vs[tt][d], f, accV[j]);
      }
    }
    __syncthreads();
  }
#pragma unroll
  for (int j = 0; j < 8; ++j) {
    int l = l0 + lq * 8 + j;
    kf[l * D_MOD + h * HDIM + d] = accK[j];
    vf[l * D_MOD + h * HDIM + d] = accV[j];
    kfb[l * D_MOD + h * HDIM + d] = (__bf16)accK[j];
    vfb[l * D_MOD + h * HDIM + d] = (__bf16)accV[j];
  }
}

// ----- gates: g[h,l] = relu( v_f^T (wgz.*kv) k_f + b )^2 + eps --------------
__global__ __launch_bounds__(256)
void gate_kernel(const float* __restrict__ vf, const float* __restrict__ kf,
                 const float* __restrict__ wgz, const float* __restrict__ kvs,
                 const float* __restrict__ wgz_b, float* __restrict__ gates) {
  __shared__ float Wg[64][64];
  int h = blockIdx.y;
  int l = blockIdx.x * 256 + threadIdx.x;
  for (int i = threadIdx.x; i < 4096; i += 256)
    Wg[i >> 6][i & 63] = wgz[i] * kvs[i];
  __syncthreads();
  const float* vp = vf + l * D_MOD + h * HDIM;
  const float* kp = kf + l * D_MOD + h * HDIM;
  float kr[64];
#pragma unroll
  for (int e = 0; e < 64; ++e) kr[e] = kp[e];
  float logit = 0.0f;
  for (int d = 0; d < 64; ++d) {
    float ud = 0.0f;
#pragma unroll 8
    for (int e = 0; e < 64; ++e) ud = fmaf(Wg[d][e], kr[e], ud);
    logit = fmaf(vp[d], ud, logit);
  }
  logit += wgz_b[0];
  float rl = fmaxf(logit, 0.0f);
  gates[h * L_SEQ + l] = rl * rl + EPSF;
}

// ----- causal gated attention: sp = l2norm( (g .* (Q Vf^T)) Kf ) ------------
// block = 128 threads = 4 waves; each wave owns 16 Q rows (64 rows / block).
// V_f / K_f tiles arrive as bf16 via async direct-to-LDS copies.
__global__ __launch_bounds__(128)
void attn_kernel(const float* __restrict__ q, const __bf16* __restrict__ vfb,
                 const __bf16* __restrict__ kfb, const float* __restrict__ gates,
                 __bf16* __restrict__ sp) {
  __shared__ __bf16 vfs[32][72];        // [t][d], padded rows
  __shared__ __bf16 kfs[32][72];        // [t][e], padded rows
  __shared__ float  gs[32];
  __shared__ __bf16 ss[4][16][32];      // per-wave gated score scratch
  int h = blockIdx.y;
  int l0 = blockIdx.x * 64;
  int wave = threadIdx.x >> 5, lane = threadIdx.x & 31;
  int m0 = l0 + wave * 16;
  int r = lane & 15, half = lane >> 4;

  // Q tile in WMMA A layout, two K-chunks over d
  v16bf aq[2];
  {
    const float* qp = q + (m0 + r) * D_MOD + h * HDIM;
#pragma unroll
    for (int kc = 0; kc < 2; ++kc) {
      int cbase = kc * 32 + half * 8;
      v16bf a;
#pragma unroll
      for (int i = 0; i < 8; ++i) a[i] = (__bf16)qp[cbase + i];
#pragma unroll
      for (int i = 0; i < 8; ++i) a[8 + i] = (__bf16)qp[cbase + 16 + i];
      aq[kc] = a;
    }
  }

  v8f oacc[4] = {};
  int tmax = l0 + 63;
  for (int t0 = 0; t0 <= tmax; t0 += 32) {
    // async tile fills: 32 rows x 64 bf16 = 256 x 16B chunks per tile
    for (int c = threadIdx.x; c < 256; c += 128) {
      int row = c >> 3, ch = (c & 7) * 8;
      cp16B(vfb + (t0 + row) * D_MOD + h * HDIM + ch, &vfs[row][ch]);
      cp16B(kfb + (t0 + row) * D_MOD + h * HDIM + ch, &kfs[row][ch]);
    }
    if (threadIdx.x < 32) gs[threadIdx.x] = gates[h * L_SEQ + t0 + threadIdx.x];
    ASYNC_WAIT(0);
    __syncthreads();

    bool active = (t0 <= m0 + 15);
    if (active) {
      // S[16 x 32] = Q . Vf^T   (B[d,t] = vfs[t][d]; contiguous LDS rows)
      v8f sacc[2];
#pragma unroll
      for (int nh = 0; nh < 2; ++nh) {
        v8f s = {};
#pragma unroll
        for (int kc = 0; kc < 2; ++kc) {
          v16bf b;
          int br = 16 * nh + r;
          int bc = kc * 32 + half * 8;
#pragma unroll
          for (int i = 0; i < 16; ++i) b[i] = vfs[br][bc + (i < 8 ? i : i + 8)];
          s = __builtin_amdgcn_wmma_f32_16x16x32_bf16(
              false, aq[kc], false, b, (short)0, s, false, false);
        }
        sacc[nh] = s;
      }
      // causal mask + gate, park as bf16 A-matrix source
#pragma unroll
      for (int nh = 0; nh < 2; ++nh) {
        int tloc = nh * 16 + r;
        int tglob = t0 + tloc;
        float g = gs[tloc];
#pragma unroll
        for (int j = 0; j < 8; ++j) {
          int mrow = m0 + j + half * 8;
          float val = (tglob <= mrow) ? sacc[nh][j] * g : 0.0f;
          ss[wave][j + half * 8][tloc] = (__bf16)val;
        }
      }
    }
    __syncthreads();
    if (active) {
      // A2 = gated S in A layout; O += A2 . Kf   (B[t,e] = kfs[t][e])
      v16bf a2;
      {
        int c = half * 8;
#pragma unroll
        for (int i = 0; i < 8; ++i) a2[i] = ss[wave][r][c + i];
#pragma unroll
        for (int i = 0; i < 8; ++i) a2[8 + i] = ss[wave][r][c + 16 + i];
      }
#pragma unroll
      for (int n = 0; n < 4; ++n) {
        v16bf b;
        int col = 16 * n + r;
        int tb = half * 8;
#pragma unroll
        for (int i = 0; i < 16; ++i) b[i] = kfs[tb + (i < 8 ? i : i + 8)][col];
        oacc[n] = __builtin_amdgcn_wmma_f32_16x16x32_bf16(
            false, a2, false, b, (short)0, oacc[n], false, false);
      }
    }
    __syncthreads();
  }

  // l2norm over e per output row, then store sp as bf16 [L, D]
  float nrm2[8] = {};
#pragma unroll
  for (int n = 0; n < 4; ++n)
#pragma unroll
    for (int j = 0; j < 8; ++j) nrm2[j] = fmaf(oacc[n][j], oacc[n][j], nrm2[j]);
#pragma unroll
  for (int j = 0; j < 8; ++j) {
    float s = nrm2[j];
#pragma unroll
    for (int m = 1; m <= 8; m <<= 1) s += __shfl_xor(s, m, 32);  // 16-lane halves
    nrm2[j] = 1.0f / fmaxf(sqrtf(s), EPSF);
  }
#pragma unroll
  for (int n = 0; n < 4; ++n) {
    int e = h * HDIM + 16 * n + r;
#pragma unroll
    for (int j = 0; j < 8; ++j) {
      int row = m0 + j + half * 8;
      sp[row * D_MOD + e] = (__bf16)(oacc[n][j] * nrm2[j]);
    }
  }
}

// ---------------------------------------------------------------------------
extern "C" void kernel_launch(void* const* d_in, const int* in_sizes, int n_in,
                              void* d_out, int out_size, void* d_ws, size_t ws_size,
                              hipStream_t stream) {
  (void)in_sizes; (void)n_in; (void)out_size; (void)ws_size;
  const float* x     = (const float*)d_in[0];
  const float* sb    = (const float*)d_in[1];
  const float* wq_w  = (const float*)d_in[2];
  const float* wq_b  = (const float*)d_in[3];
  const float* wk_w  = (const float*)d_in[4];
  const float* wk_b  = (const float*)d_in[5];
  const float* wv_w  = (const float*)d_in[6];
  const float* wv_b  = (const float*)d_in[7];
  const float* wo_w  = (const float*)d_in[8];
  const float* wo_b  = (const float*)d_in[9];
  const float* td_w  = (const float*)d_in[10];
  const float* td_b  = (const float*)d_in[11];
  const float* wgz_w = (const float*)d_in[12];
  const float* wgz_b = (const float*)d_in[13];
  const float* kv_s  = (const float*)d_in[14];
  // d_in[15] qk_norm_scale: unused by the reference output
  float* out = (float*)d_out;

  char* ws = (char*)d_ws;
  size_t off = 0;
  auto alloc = [&](size_t bytes) -> void* {
    void* p = ws + off;
    off = (off + bytes + 255) & ~(size_t)255;
    return p;
  };
  const int LD = L_SEQ * D_MOD;
  const int DD = D_MOD * D_MOD;
  __bf16* xb   = (__bf16*)alloc(LD * 2);
  __bf16* sbb  = (__bf16*)alloc(LD * 2);
  __bf16* wqb  = (__bf16*)alloc(DD * 2);
  __bf16* wkb  = (__bf16*)alloc(DD * 2);
  __bf16* wvb  = (__bf16*)alloc(DD * 2);
  __bf16* tdb  = (__bf16*)alloc(DD * 2);
  __bf16* wob  = (__bf16*)alloc(DD * 2);
  float*  qf   = (float*)alloc(LD * 4);
  float*  kraw = (float*)alloc(LD * 4);
  float*  vraw = (float*)alloc(LD * 4);
  float*  fltf = (float*)alloc(LD * 4);
  float*  kn   = (float*)alloc(LD * 4);
  float*  vn   = (float*)alloc(LD * 4);
  float*  kfc  = (float*)alloc(LD * 4);
  float*  vfc  = (float*)alloc(LD * 4);
  __bf16* kfb  = (__bf16*)alloc(LD * 2);
  __bf16* vfb  = (__bf16*)alloc(LD * 2);
  float*  gts  = (float*)alloc(NHEAD * L_SEQ * 4);
  __bf16* spb  = (__bf16*)alloc(LD * 2);

  // 1) fp32 -> bf16 conversions
  cvt_f32_bf16<<<LD / 256, 256, 0, stream>>>(x,    xb,  LD);
  cvt_f32_bf16<<<LD / 256, 256, 0, stream>>>(sb,   sbb, LD);
  cvt_f32_bf16<<<DD / 256, 256, 0, stream>>>(wq_w, wqb, DD);
  cvt_f32_bf16<<<DD / 256, 256, 0, stream>>>(wk_w, wkb, DD);
  cvt_f32_bf16<<<DD / 256, 256, 0, stream>>>(wv_w, wvb, DD);
  cvt_f32_bf16<<<DD / 256, 256, 0, stream>>>(td_w, tdb, DD);
  cvt_f32_bf16<<<DD / 256, 256, 0, stream>>>(wo_w, wob, DD);

  // 2) projections (WMMA + async-LDS): grid (N/128, M/64)
  dim3 ggrid(D_MOD / 128, L_SEQ / 64);
  gemm_bf16_wmma<<<ggrid, 256, 0, stream>>>(xb,  wqb, wq_b, qf,   L_SEQ, D_MOD, D_MOD);
  gemm_bf16_wmma<<<ggrid, 256, 0, stream>>>(xb,  wkb, wk_b, kraw, L_SEQ, D_MOD, D_MOD);
  gemm_bf16_wmma<<<ggrid, 256, 0, stream>>>(xb,  wvb, wv_b, vraw, L_SEQ, D_MOD, D_MOD);
  gemm_bf16_wmma<<<ggrid, 256, 0, stream>>>(sbb, tdb, td_b, fltf, L_SEQ, D_MOD, D_MOD);

  // 3) per-head L2 norms of k, v
  l2norm_rows<<<(L_SEQ * NHEAD) / 8, 256, 0, stream>>>(kraw, vraw, kn, vn);

  // 4) causal convolutions with spectral filters (fp32 + bf16 outputs)
  conv_causal<<<dim3(L_SEQ / 32, NHEAD), 256, 0, stream>>>(kn, vn, fltf, kfc, vfc, kfb, vfb);

  // 5) gates
  gate_kernel<<<dim3(L_SEQ / 256, NHEAD), 256, 0, stream>>>(vfc, kfc, wgz_w, kv_s, wgz_b, gts);

  // 6) causal gated attention + l2norm -> sp (bf16)
  attn_kernel<<<dim3(L_SEQ / 64, NHEAD), 128, 0, stream>>>(qf, vfb, kfb, gts, spb);

  // 7) output projection (WMMA) -> d_out (fp32)
  gemm_bf16_wmma<<<ggrid, 256, 0, stream>>>(spb, wob, wo_b, out, L_SEQ, D_MOD, D_MOD);
}
